// PatchLocalPoolPointnetLatent_40570261078477
// MI455X (gfx1250) — matrix operand order repaired and checked by reading, not compiled, not run on hardware
//
#include <hip/hip_runtime.h>
#include <hip/hip_bf16.h>

// ---------------- problem constants (from reference) ----------------
#define RESO     32
#define NSEG     65536        // G = 2*RESO^3
#define HID      128
#define CDIM     128
#define NBLK     5
#define NB       2            // batch
#define NPTS     100000       // points per batch  (divisible by 32)
#define TILE_M   32                     // points per wave
#define TILES    ((NB*NPTS)/TILE_M)     // 6250 tiles
#define WG_TILES 2                      // waves (tiles) per workgroup
#define NWG      (TILES/WG_TILES)       // 3125
#define XSTR     264                    // LDS x-tile row stride (halves)
#define HSTR     136                    // LDS h-tile row stride (halves)

typedef __attribute__((ext_vector_type(16))) _Float16 v16h;
typedef __attribute__((ext_vector_type(8)))  _Float16 v8h;
typedef __attribute__((ext_vector_type(2)))  _Float16 v2h;
typedef __attribute__((ext_vector_type(8)))  float    v8f;
typedef __attribute__((ext_vector_type(8)))  unsigned int v8u;

// ------------------------------------------------------------------
// helpers
// ------------------------------------------------------------------
__device__ __forceinline__ v8f wmma_f16(v16h a, v16h b, v8f c) {
    // D = A(16x32 f16) * B(32x16 f16) + C(16x16 f32)
    return __builtin_amdgcn_wmma_f32_16x16x32_f16(
        /*neg_a=*/false, a, /*neg_b=*/false, b,
        /*c_mod=*/(short)0, c, /*reuse_a=*/false, /*reuse_b=*/false);
}

// A-fragment from an f16 row in LDS.
// ISA layout (16-bit A, 16x32): lanes 0-15 hold K = {kb..kb+7, kb+16..kb+23}
// with kb = kc*32; lanes 16-31 the same shifted by 8.
__device__ __forceinline__ v16h packA_lds(const _Float16* hrow, int kb) {
    v8h lo = *(const v8h*)(hrow + kb);
    v8h hi = *(const v8h*)(hrow + kb + 16);
    v16h r;
#pragma unroll
    for (int t = 0; t < 8; ++t) { r[t] = lo[t]; r[8 + t] = hi[t]; }
    return r;
}

// ReLU on a packed f16 fragment: exactly one v_pk_max_num_f16 per pair.
__device__ __forceinline__ v16h relu16(v16h r) {
    v8u w = __builtin_bit_cast(v8u, r);
#pragma unroll
    for (int i = 0; i < 8; ++i) {
        unsigned int o;
        asm("v_pk_max_num_f16 %0, %1, 0" : "=v"(o) : "v"(w[i]));
        w[i] = o;
    }
    return __builtin_bit_cast(v16h, w);
}

__device__ __forceinline__ v16h packA_lds_relu(const _Float16* hrow, int kb) {
    return relu16(packA_lds(hrow, kb));
}

// A-fragment for this lane from an f32 row in global memory (final_kernel).
__device__ __forceinline__ v16h packA_f32(const float* __restrict__ base,
                                          int kb) {
    const float4* p0 = (const float4*)(base + kb);
    const float4* p1 = (const float4*)(base + kb + 16);
    float4 a0 = p0[0], a1 = p0[1];
    float4 b0 = p1[0], b1 = p1[1];
    float f[16] = { a0.x, a0.y, a0.z, a0.w, a1.x, a1.y, a1.z, a1.w,
                    b0.x, b0.y, b0.z, b0.w, b1.x, b1.y, b1.z, b1.w };
    v16h r;
#pragma unroll
    for (int t = 0; t < 16; ++t) r[t] = (_Float16)f[t];
    return r;
}

// Pre-tiled B fragment: contiguous 32 bytes per lane.
__device__ __forceinline__ v16h loadB(const _Float16* __restrict__ wt,
                                      int kc, int nt, int lane) {
    return *(const v16h*)(wt + (size_t)((((kc << 3) + nt) << 5) + lane) * 16);
}

// convert 8 consecutive f32 to a 16-byte f16 chunk
__device__ __forceinline__ v8h cvt8(const float* __restrict__ src) {
    float4 v0 = *(const float4*)(src);
    float4 v1 = *(const float4*)(src + 4);
    v8h h;
    h[0] = (_Float16)v0.x; h[1] = (_Float16)v0.y;
    h[2] = (_Float16)v0.z; h[3] = (_Float16)v0.w;
    h[4] = (_Float16)v1.x; h[5] = (_Float16)v1.y;
    h[6] = (_Float16)v1.z; h[7] = (_Float16)v1.w;
    return h;
}

// packed f16x2 global atomic add (GLOBAL_ATOMIC_PK_ADD_F16, no return)
__device__ __forceinline__ void atomic_pk_add_f16(_Float16* addr, v2h val) {
    unsigned int d = __builtin_bit_cast(unsigned int, val);
    asm volatile("global_atomic_pk_add_f16 %0, %1, off"
                 :: "v"(addr), "v"(d) : "memory");
}

// ------------------------------------------------------------------
// weight pre-tiling: f32 row-major (K x N) -> f16 WMMA-B fragments.
// ------------------------------------------------------------------
__global__ void tile_weights_kernel(const float* __restrict__ src,
                                    _Float16* __restrict__ dst,
                                    int K, int N) {
    int o = blockIdx.x * 256 + threadIdx.x;
    if (o >= K * N) return;
    int t    = o & 15;
    int lane = (o >> 4) & 31;
    int ntk  = o >> 9;
    int NT   = N >> 4;
    int nt   = ntk % NT;
    int kc   = ntk / NT;
    int kl   = ((t < 8) ? t : (t + 8)) + ((lane < 16) ? 0 : 8);
    int k    = kc * 32 + kl;
    int col  = nt * 16 + (lane & 15);
    dst[o] = (_Float16)src[k * N + col];
}

// ------------------------------------------------------------------
// utility kernels
// ------------------------------------------------------------------
__global__ void zero_kernel(float* __restrict__ p, size_t n) {
    size_t i = (size_t)blockIdx.x * blockDim.x + threadIdx.x;
    if (i < n) p[i] = 0.0f;
}

__global__ void count_kernel(const long long* __restrict__ idx,
                             float* __restrict__ cnt) {
    int i = blockIdx.x * 256 + threadIdx.x;
    if (i >= NB * NPTS) return;
    int b = i / NPTS;
    long long g = idx[i];
    atomicAdd(&cnt[(size_t)b * NSEG + (size_t)g], 1.0f);
}

// scatter-add net rows into f16 grid[b][g][c] with packed-f16 atomics
__global__ void scatter_kernel(const float* __restrict__ net,
                               const long long* __restrict__ idx,
                               _Float16* __restrict__ grid) {
    size_t t = (size_t)blockIdx.x * 256 + threadIdx.x;
    if (t >= (size_t)NB * NPTS * 32) return;
    size_t pt = t >> 5;
    int c4 = (int)(t & 31) * 4;
    int b = (int)(pt / NPTS);
    long long g = idx[pt];
    float4 v = *(const float4*)(net + pt * HID + c4);
    v2h p0; p0[0] = (_Float16)v.x; p0[1] = (_Float16)v.y;
    v2h p1; p1[0] = (_Float16)v.z; p1[1] = (_Float16)v.w;
    _Float16* dst = grid + ((size_t)b * NSEG + (size_t)g) * HID + c4;
    atomic_pk_add_f16(dst + 0, p0);
    atomic_pk_add_f16(dst + 2, p1);
}

// f16 grid[b][g][c] /= max(cnt[b][g],1), one f16x2 pair per thread
__global__ void divmean_f16_kernel(const float* __restrict__ cnt,
                                   v2h* __restrict__ grid) {
    size_t i = (size_t)blockIdx.x * 256 + threadIdx.x;
    if (i >= (size_t)NB * NSEG * 64) return;     // 64 pairs per cell
    size_t bg = i >> 6;
    float inv = 1.0f / fmaxf(cnt[bg], 1.0f);
    v2h h = grid[i];
    h[0] = (_Float16)((float)h[0] * inv);
    h[1] = (_Float16)((float)h[1] * inv);
    grid[i] = h;
}

// out[b][ch][g] /= max(cnt[b][g],1)
__global__ void final_div_kernel(const float* __restrict__ cnt,
                                 float* __restrict__ out) {
    size_t i = (size_t)blockIdx.x * 256 + threadIdx.x;
    if (i >= (size_t)NB * CDIM * NSEG) return;
    size_t g = i & (NSEG - 1);
    size_t b = i >> 23;                   // 128*65536 = 2^23
    out[i] /= fmaxf(cnt[b * NSEG + g], 1.0f);
}

// ------------------------------------------------------------------
// shared resnet-block core: x-tile (32 x 256 f16) already staged in LDS.
// GEMM1: h = relu(x)@w0 (+b0, relu) -> LDS h-tile
// GEMM2: dx = h@w1 ; GEMM3: += x@ws ; epilogue +b1 -> net rows.
// ------------------------------------------------------------------
__device__ __forceinline__ void resnet_core(
        const _Float16 (*xs)[XSTR], _Float16 (*hs)[HSTR],
        float* __restrict__ net, int pt0, int lane, int row,
        const _Float16* __restrict__ w0t, const float* __restrict__ b0,
        const _Float16* __restrict__ w1t, const float* __restrict__ b1,
        const _Float16* __restrict__ wst) {
    const _Float16* xrow0 = xs[row];
    const _Float16* xrow1 = xs[16 + row];

    // ---- h = relu(x) @ w0 (+ b0 in epilogue) ----
    v8f acc[2][8] = {};
#pragma unroll
    for (int kc = 0; kc < 8; ++kc) {
        int kb = kc * 32 + ((lane < 16) ? 0 : 8);
        v16h a0 = packA_lds_relu(xrow0, kb);
        v16h a1 = packA_lds_relu(xrow1, kb);
#pragma unroll
        for (int nt = 0; nt < 8; ++nt) {
            v16h bb = loadB(w0t, kc, nt, lane);
            acc[0][nt] = wmma_f16(a0, bb, acc[0][nt]);
            acc[1][nt] = wmma_f16(a1, bb, acc[1][nt]);
        }
    }
    float bv0[8];
#pragma unroll
    for (int nt = 0; nt < 8; ++nt) bv0[nt] = b0[nt * 16 + row];
#pragma unroll
    for (int g2 = 0; g2 < 2; ++g2)
#pragma unroll
        for (int nt = 0; nt < 8; ++nt)
#pragma unroll
            for (int r = 0; r < 8; ++r) {
                int m = g2 * 16 + r + ((lane < 16) ? 0 : 8);
                hs[m][nt * 16 + row] =
                    (_Float16)fmaxf(acc[g2][nt][r] + bv0[nt], 0.0f);
            }
    __syncthreads();

    // ---- dx = h @ w1 ; then += x @ ws ; (+ b1 in epilogue) ----
    v8f acc2[2][8] = {};
    const _Float16* hrow0 = hs[row];
    const _Float16* hrow1 = hs[16 + row];
#pragma unroll
    for (int kc = 0; kc < 4; ++kc) {
        int kb = kc * 32 + ((lane < 16) ? 0 : 8);
        v16h a0 = packA_lds(hrow0, kb);
        v16h a1 = packA_lds(hrow1, kb);
#pragma unroll
        for (int nt = 0; nt < 8; ++nt) {
            v16h bb = loadB(w1t, kc, nt, lane);
            acc2[0][nt] = wmma_f16(a0, bb, acc2[0][nt]);
            acc2[1][nt] = wmma_f16(a1, bb, acc2[1][nt]);
        }
    }
#pragma unroll
    for (int kc = 0; kc < 8; ++kc) {
        int kb = kc * 32 + ((lane < 16) ? 0 : 8);
        v16h a0 = packA_lds(xrow0, kb);
        v16h a1 = packA_lds(xrow1, kb);
#pragma unroll
        for (int nt = 0; nt < 8; ++nt) {
            v16h bb = loadB(wst, kc, nt, lane);
            acc2[0][nt] = wmma_f16(a0, bb, acc2[0][nt]);
            acc2[1][nt] = wmma_f16(a1, bb, acc2[1][nt]);
        }
    }
    float bv1[8];
#pragma unroll
    for (int nt = 0; nt < 8; ++nt) bv1[nt] = b1[nt * 16 + row];
#pragma unroll
    for (int g2 = 0; g2 < 2; ++g2)
#pragma unroll
        for (int r = 0; r < 8; ++r) {
            int m = g2 * 16 + r + ((lane < 16) ? 0 : 8);
            float* orow = net + (size_t)(pt0 + m) * HID;
#pragma unroll
            for (int nt = 0; nt < 8; ++nt)
                orow[nt * 16 + row] = acc2[g2][nt][r] + bv1[nt];
        }
}

// ------------------------------------------------------------------
// Block 0: x = fc_pos(points), computed once into LDS (one row per lane),
// then the resnet core.
// ------------------------------------------------------------------
__global__ __launch_bounds__(64) void block0_kernel(
        const float* __restrict__ points, float* __restrict__ net,
        const float* __restrict__ wpos, const float* __restrict__ bpos,
        const _Float16* __restrict__ w0t, const float* __restrict__ b0,
        const _Float16* __restrict__ w1t, const float* __restrict__ b1,
        const _Float16* __restrict__ wst) {
    __shared__ __align__(16) _Float16 xsh[WG_TILES][TILE_M][XSTR];
    __shared__ __align__(16) _Float16 hsh[WG_TILES][TILE_M][HSTR];
    const int lane = threadIdx.x & 31;
    const int wid  = threadIdx.x >> 5;
    const int pt0  = (blockIdx.x * WG_TILES + wid) * TILE_M;
    const int row  = lane & 15;

    __builtin_prefetch(w0t, 0, 1);
    __builtin_prefetch(wst, 0, 1);

    // stage x = fc_pos(point) for my row (one row per lane)
    {
        const float* prow = points + (size_t)(pt0 + lane) * 3;
        float px = prow[0], py = prow[1], pz = prow[2];
        _Float16* xrow = xsh[wid][lane];
#pragma unroll
        for (int c = 0; c < 256; c += 8) {
            v8h h;
#pragma unroll
            for (int j = 0; j < 8; ++j) {
                int cc = c + j;
                float v = fmaf(px, wpos[cc],
                          fmaf(py, wpos[256 + cc],
                          fmaf(pz, wpos[512 + cc], bpos[cc])));
                h[j] = (_Float16)v;
            }
            *(v8h*)(xrow + c) = h;
        }
    }
    __syncthreads();

    resnet_core(xsh[wid], hsh[wid], net, pt0, lane, row,
                w0t, b0, w1t, b1, wst);
}

// ------------------------------------------------------------------
// Blocks 1..4: x = concat(net[128], pooled[128]) staged into LDS (one row
// per lane). pooled grid is f16 -> copied straight into LDS, no converts.
// In-place net update is safe: each wave reads only its own rows before
// storing them.
// ------------------------------------------------------------------
__global__ __launch_bounds__(64) void resblock_kernel(
        float* __restrict__ net, const _Float16* __restrict__ pooled,
        const long long* __restrict__ idx,
        const _Float16* __restrict__ w0t, const float* __restrict__ b0,
        const _Float16* __restrict__ w1t, const float* __restrict__ b1,
        const _Float16* __restrict__ wst) {
    __shared__ __align__(16) _Float16 xsh[WG_TILES][TILE_M][XSTR];
    __shared__ __align__(16) _Float16 hsh[WG_TILES][TILE_M][HSTR];
    const int lane = threadIdx.x & 31;
    const int wid  = threadIdx.x >> 5;
    const int pt0  = (blockIdx.x * WG_TILES + wid) * TILE_M;
    const int b    = pt0 / NPTS;
    const int n0   = pt0 - b * NPTS;
    const int row  = lane & 15;

    __builtin_prefetch(w0t, 0, 1);
    __builtin_prefetch(wst, 0, 1);

    // stage x = [net row | pooled row] (one row per lane)
    {
        long long g = idx[(size_t)b * NPTS + n0 + lane];
        const float* nrow = net + (size_t)(pt0 + lane) * HID;
        const _Float16* prow = pooled + ((size_t)b * NSEG + (size_t)g) * HID;
        _Float16* xrow = xsh[wid][lane];
#pragma unroll
        for (int c = 0; c < 128; c += 8)
            *(v8h*)(xrow + c) = cvt8(nrow + c);
#pragma unroll
        for (int c = 0; c < 128; c += 8)
            *(v8h*)(xrow + 128 + c) = *(const v8h*)(prow + c);
    }
    __syncthreads();

    resnet_core(xsh[wid], hsh[wid], net, pt0, lane, row,
                w0t, b0, w1t, b1, wst);
}

// ------------------------------------------------------------------
// Final: c = net @ fc_c_w + fc_c_b, scatter-add into out[b][ch][g] (f32).
// ------------------------------------------------------------------
__global__ __launch_bounds__(64) void final_kernel(
        const float* __restrict__ net, const long long* __restrict__ idx,
        const _Float16* __restrict__ wct, const float* __restrict__ bc,
        float* __restrict__ out) {
    const int lane = threadIdx.x & 31;
    const int wid  = threadIdx.x >> 5;
    const int pt0  = (blockIdx.x * WG_TILES + wid) * TILE_M;
    const int b    = pt0 / NPTS;
    const int n0   = pt0 - b * NPTS;
    const int row  = lane & 15;

    __builtin_prefetch(wct, 0, 1);

    const float* netrow0 = net + (size_t)(pt0 + row) * HID;
    const float* netrow1 = net + (size_t)(pt0 + 16 + row) * HID;
    v8f acc[2][8] = {};
#pragma unroll
    for (int kc = 0; kc < 4; ++kc) {
        int kb = kc * 32 + ((lane < 16) ? 0 : 8);
        v16h a0 = packA_f32(netrow0, kb);
        v16h a1 = packA_f32(netrow1, kb);
#pragma unroll
        for (int nt = 0; nt < 8; ++nt) {
            v16h bb = loadB(wct, kc, nt, lane);
            acc[0][nt] = wmma_f16(a0, bb, acc[0][nt]);
            acc[1][nt] = wmma_f16(a1, bb, acc[1][nt]);
        }
    }
    float bv[8];
#pragma unroll
    for (int nt = 0; nt < 8; ++nt) bv[nt] = bc[nt * 16 + row];
#pragma unroll
    for (int g2 = 0; g2 < 2; ++g2)
#pragma unroll
        for (int r = 0; r < 8; ++r) {
            int m = g2 * 16 + r + ((lane < 16) ? 0 : 8);
            long long gm = idx[(size_t)b * NPTS + n0 + m];
            float* obase = out + (size_t)b * CDIM * NSEG + (size_t)gm;
#pragma unroll
            for (int nt = 0; nt < 8; ++nt)
                atomicAdd(obase + (size_t)(nt * 16 + row) * NSEG,
                          acc[g2][nt][r] + bv[nt]);
        }
}

// ------------------------------------------------------------------
// host-side launcher
// ------------------------------------------------------------------
extern "C" void kernel_launch(void* const* d_in, const int* in_sizes, int n_in,
                              void* d_out, int out_size, void* d_ws, size_t ws_size,
                              hipStream_t stream) {
    const float*     points   = (const float*)d_in[0];
    const long long* index    = (const long long*)d_in[1];
    const float*     fc_pos_w = (const float*)d_in[2];
    const float*     fc_pos_b = (const float*)d_in[3];
    const float*     fc0_w    = (const float*)d_in[4];   // (5,256,128)
    const float*     fc0_b    = (const float*)d_in[5];   // (5,128)
    const float*     fc1_w    = (const float*)d_in[6];   // (5,128,128)
    const float*     fc1_b    = (const float*)d_in[7];   // (5,128)
    const float*     sc_w     = (const float*)d_in[8];   // (5,256,128)
    const float*     fc_c_w   = (const float*)d_in[9];   // (128,128)
    const float*     fc_c_b   = (const float*)d_in[10];  // (128,)
    float* out = (float*)d_out;                          // B*128*65536 f32

    // workspace layout
    char* ws = (char*)d_ws;
    float* net = (float*)ws;                                   // B*N*128 f32
    size_t off = (size_t)NB * NPTS * HID * sizeof(float);
    float* cnt = (float*)(ws + off); off += (size_t)NB * NSEG * sizeof(float);
    _Float16* w0t = (_Float16*)(ws + off); off += (size_t)NBLK * 256 * 128 * 2;
    _Float16* w1t = (_Float16*)(ws + off); off += (size_t)NBLK * 128 * 128 * 2;
    _Float16* wst = (_Float16*)(ws + off); off += (size_t)NBLK * 256 * 128 * 2;
    _Float16* wct = (_Float16*)(ws + off); off += (size_t)128 * 128 * 2;

    // pre-tile weights to WMMA B-fragment f16 layout
    for (int i = 0; i < NBLK; ++i) {
        tile_weights_kernel<<<(256 * 128 + 255) / 256, 256, 0, stream>>>(
            fc0_w + (size_t)i * 256 * 128, w0t + (size_t)i * 256 * 128, 256, 128);
        tile_weights_kernel<<<(128 * 128 + 255) / 256, 256, 0, stream>>>(
            fc1_w + (size_t)i * 128 * 128, w1t + (size_t)i * 128 * 128, 128, 128);
        tile_weights_kernel<<<(256 * 128 + 255) / 256, 256, 0, stream>>>(
            sc_w + (size_t)i * 256 * 128, wst + (size_t)i * 256 * 128, 256, 128);
    }
    tile_weights_kernel<<<(128 * 128 + 255) / 256, 256, 0, stream>>>(fc_c_w, wct, 128, 128);

    // segment counts (idx is constant across the pipeline)
    zero_kernel<<<(NB * NSEG + 255) / 256, 256, 0, stream>>>(cnt, (size_t)NB * NSEG);
    count_kernel<<<(NB * NPTS + 255) / 256, 256, 0, stream>>>(index, cnt);

    // block 0 (fc_pos fused)
    block0_kernel<<<NWG, 64, 0, stream>>>(points, net, fc_pos_w, fc_pos_b,
                                          w0t, fc0_b, w1t, fc1_b, wst);

    // d_out is reused as the intermediate f16 pooled grid [b][g][c]
    _Float16* gridh = (_Float16*)d_out;
    const size_t gridH32 = (size_t)NB * NSEG * 64;   // f16 grid in 32-bit words
    const size_t gridN   = (size_t)NB * NSEG * HID;  // f32 output elements
    for (int i = 1; i < NBLK; ++i) {
        zero_kernel<<<(unsigned)((gridH32 + 255) / 256), 256, 0, stream>>>(
            (float*)gridh, gridH32);
        scatter_kernel<<<(unsigned)(((size_t)NB * NPTS * 32 + 255) / 256), 256, 0, stream>>>(
            net, index, gridh);
        divmean_f16_kernel<<<(unsigned)((gridH32 + 255) / 256), 256, 0, stream>>>(
            cnt, (v2h*)gridh);
        resblock_kernel<<<NWG, 64, 0, stream>>>(
            net, gridh, index,
            w0t + (size_t)i * 256 * 128, fc0_b + (size_t)i * 128,
            w1t + (size_t)i * 128 * 128, fc1_b + (size_t)i * 128,
            wst + (size_t)i * 256 * 128);
    }

    // final projection + segment mean into transposed output (f32)
    zero_kernel<<<(unsigned)((gridN + 255) / 256), 256, 0, stream>>>(out, gridN);
    final_kernel<<<NWG, 64, 0, stream>>>(net, index, wct, fc_c_b, out);
    final_div_kernel<<<(unsigned)((gridN + 255) / 256), 256, 0, stream>>>(cnt, out);
}
